// SelfAdaptivePreisachModel_35038343201619
// MI455X (gfx1250) — matrix operand order repaired and checked by reading, not compiled, not run on hardware
//
#include <hip/hip_runtime.h>
#include <hip/hip_bf16.h>

typedef _Float16 v16h __attribute__((ext_vector_type(16)));
typedef float    v8f  __attribute__((ext_vector_type(8)));

namespace {
constexpr int   kT       = 4096;
constexpr int   kN       = 20301;               // 201*202/2 Preisach triangle
constexpr float kInvTemp = 1000.0f;             // 1/TEMP
constexpr float kLog2e   = 1.4426950408889634f;

__device__ __forceinline__ float fast_sigmoid(float z) {
  // sigmoid(z) = 1 / (1 + 2^(-z*log2e)) via v_exp_f32 + v_rcp_f32
  return __builtin_amdgcn_rcpf(1.0f + __builtin_amdgcn_exp2f(-kLog2e * z));
}
}  // namespace

// ---------------------------------------------------------------------------
// Kernel 1: density MLP. One wave (32 threads) per 16-row tile of the mesh.
// Layers 2,3 (16x16, K=16 padded to 32) run on v_wmma_f32_16x16x32_f16.
//   A frag (16x32 f16): lane l holds row M=l&15; lanes 0-15 carry K=0..7,16..23,
//   lanes 16-31 carry K=8..15,24..31 (we zero K>=16).
//   B frag (32x16 f16): lane l holds col N=l&15; lanes 0-15 carry K=0..7,16..23,
//   lanes 16-31 carry K=8..15,24..31 (we zero K>=16).
//   C/D (16x16 f32): lane l holds col N=l&15; VGPR r holds row M=r+8*(l>>4).
// All loads are unconditional (clamped/valid addresses) so EXEC stays all-ones
// and the half-wave zeroing lowers to VALU selects, not exec-mask branches.
// ---------------------------------------------------------------------------
__global__ __launch_bounds__(32) void mlp_density_kernel(
    const float* __restrict__ alpha, const float* __restrict__ beta,
    const float* __restrict__ w1, const float* __restrict__ b1,
    const float* __restrict__ w2, const float* __restrict__ b2,
    const float* __restrict__ w3, const float* __restrict__ b3,
    const float* __restrict__ w4, const float* __restrict__ b4,
    float* __restrict__ density) {
  __shared__ float hb[16 * 16];
  const int lane = threadIdx.x;
  const int m    = lane & 15;   // row for A/layer1, col N for B/C
  const int half = lane >> 4;
  const float sel = (half == 0) ? 1.0f : 0.0f;  // branchless half-wave select
  const int row  = blockIdx.x * 16 + m;
  const int rowc = row < kN ? row : kN - 1;     // clamp: keep EXEC all-ones
  const float a  = alpha[rowc];
  const float bb = beta[rowc];

  // Layer 1 (K=2): plain VALU, tile [16 rows x 16 hidden] -> LDS.
  #pragma unroll
  for (int j = 0; j < 8; ++j) {
    const int col = half * 8 + j;
    const float v = fmaf(a, w1[col], fmaf(bb, w1[16 + col], b1[col]));
    hb[m * 16 + col] = fmaxf(v, 0.0f);
  }
  __syncthreads();

  // Layers 2 and 3 on WMMA.
  const float* wsel[2] = {w2, w3};
  const float* bsel[2] = {b2, b3};
  #pragma unroll
  for (int layer = 0; layer < 2; ++layer) {
    const float* w  = wsel[layer];
    const float  bn = bsel[layer][m];
    v16h af, bf;
    #pragma unroll
    for (int i = 0; i < 8; ++i) {
      af[i]     = (_Float16)hb[m * 16 + half * 8 + i];
      af[i + 8] = (_Float16)0.0f;                 // K = 16..31 zero pad
    }
    #pragma unroll
    for (int j = 0; j < 16; ++j) {
      // Unconditional, coalesced load; K = half*8 + j for j < 8, zero for j >= 8.
      bf[j] = (_Float16)(w[(half * 8 + (j & 7)) * 16 + m] * (j < 8 ? 1.0f : 0.0f));
    }
    v8f acc = {};
    acc = __builtin_amdgcn_wmma_f32_16x16x32_f16(false, af, false, bf,
                                                 (short)0, acc, false, false);
    __syncthreads();
    #pragma unroll
    for (int r = 0; r < 8; ++r)
      hb[(half * 8 + r) * 16 + m] = fmaxf(acc[r] + bn, 0.0f);  // bias + relu
    __syncthreads();
  }

  // Layer 4 (16 -> 1) + sigmoid.
  if (half == 0) {
    float accv = b4[0];
    #pragma unroll
    for (int j = 0; j < 16; ++j) accv = fmaf(hb[m * 16 + j], w4[j], accv);
    if (row < kN) density[row] = fast_sigmoid(accv);
  }
}

// ---------------------------------------------------------------------------
// Kernel 2: deterministic fixed-tree sum of density -> ws[0].
// ---------------------------------------------------------------------------
__global__ __launch_bounds__(1024) void density_sum_kernel(
    const float* __restrict__ density, float* __restrict__ sumdens) {
  __shared__ float red[32];
  float p = 0.0f;
  for (int n = threadIdx.x; n < kN; n += 1024) p += density[n];
  #pragma unroll
  for (int d = 1; d < 32; d <<= 1) p += __shfl_xor(p, d);
  if ((threadIdx.x & 31) == 0) red[threadIdx.x >> 5] = p;
  __syncthreads();
  if (threadIdx.x == 0) {
    float total = 0.0f;
    #pragma unroll
    for (int w = 0; w < 32; ++w) total += red[w];
    sumdens[0] = total;
  }
}

// ---------------------------------------------------------------------------
// Kernel 3: the sequential Preisach scan. One lane per relay.
// s_{t+1} = c_t + g_t * s_t ; c,g depend only on x[t] -> unroll by 8 so the
// exp/rcp chains of 8 steps run with full ILP; only the 8 FMAs are serial.
// Stores are coalesced across lanes (states is [T, N]).
// ---------------------------------------------------------------------------
__global__ __launch_bounds__(256) void scan_kernel(
    const float* __restrict__ x, const float* __restrict__ alpha,
    const float* __restrict__ beta, const float* __restrict__ init_raw,
    float* __restrict__ states) {
  __shared__ float xs[kT];
  for (int i = threadIdx.x; i < kT; i += 256) xs[i] = x[i];
  __syncthreads();

  const int  n     = blockIdx.x * 256 + threadIdx.x;
  const bool valid = n < kN;
  const int  nc    = valid ? n : kN - 1;
  const float K  = kInvTemp * kLog2e;
  const float aK = alpha[nc] * K;
  const float bK = beta[nc] * K;
  float s = tanhf(init_raw[nc]);

  for (int t0 = 0; t0 < kT; t0 += 8) {
    float c[8], g[8];
    #pragma unroll
    for (int i = 0; i < 8; ++i) {
      const float uK = xs[t0 + i] * K;
      // up = sigmoid((u-a)/temp) ; dn = sigmoid((b-u)/temp)
      const float up = __builtin_amdgcn_rcpf(1.0f + __builtin_amdgcn_exp2f(aK - uK));
      const float dn = __builtin_amdgcn_rcpf(1.0f + __builtin_amdgcn_exp2f(uK - bK));
      c[i] = up - dn;
      g[i] = (1.0f - up) * (1.0f - dn);
    }
    #pragma unroll
    for (int i = 0; i < 8; ++i) {
      s = fmaf(g[i], s, c[i]);
      if (valid) states[(size_t)(t0 + i) * kN + n] = s;
    }
  }
}

// ---------------------------------------------------------------------------
// Kernel 4: m[t] = dot(density, states[t,:]) / sum_density, then affine out.
// One block per t, processed tail-first so recently written states rows hit
// the 192 MB L2. Fixed-order reduction -> bitwise deterministic.
// ---------------------------------------------------------------------------
__global__ __launch_bounds__(256) void finalize_kernel(
    const float* __restrict__ x, const float* __restrict__ density,
    const float* __restrict__ states, const float* __restrict__ sumdens,
    const float* __restrict__ m_scale_raw, const float* __restrict__ m_offset_raw,
    const float* __restrict__ h_scale_raw, float* __restrict__ out) {
  const int t = kT - 1 - (int)blockIdx.x;
  __shared__ float red[8];
  const float* srow = states + (size_t)t * kN;
  float p = 0.0f;
  #pragma unroll 4
  for (int n = threadIdx.x; n < kN; n += 256) p = fmaf(density[n], srow[n], p);
  #pragma unroll
  for (int d = 1; d < 32; d <<= 1) p += __shfl_xor(p, d);
  if ((threadIdx.x & 31) == 0) red[threadIdx.x >> 5] = p;
  __syncthreads();
  if (threadIdx.x == 0) {
    float num = 0.0f;
    #pragma unroll
    for (int w = 0; w < 8; ++w) num += red[w];
    const float m  = num / sumdens[0];
    const float ms = 10.0f * fast_sigmoid(m_scale_raw[0]);
    const float mo = fmaf(20.0f, fast_sigmoid(m_offset_raw[0]), -10.0f);
    const float hs = 10.0f * fast_sigmoid(h_scale_raw[0]);
    out[t] = fmaf(ms, m, mo) + hs * x[t];
  }
}

// ---------------------------------------------------------------------------
// Host launcher. Input order = setup_inputs() dict order:
// 0:x 1:alpha 2:beta 3:init_state 4:m_scale 5:m_offset 6:h_scale
// 7:w1 8:b1 9:w2 10:b2 11:w3 12:b3 13:w4 14:b4
// d_out = [ out(4096) | density(20301) | states(4096*20301) ]
// ---------------------------------------------------------------------------
extern "C" void kernel_launch(void* const* d_in, const int* in_sizes, int n_in,
                              void* d_out, int out_size, void* d_ws, size_t ws_size,
                              hipStream_t stream) {
  (void)in_sizes; (void)n_in; (void)out_size; (void)ws_size;
  const float* x        = (const float*)d_in[0];
  const float* alpha    = (const float*)d_in[1];
  const float* beta     = (const float*)d_in[2];
  const float* init_raw = (const float*)d_in[3];
  const float* msr      = (const float*)d_in[4];
  const float* mor      = (const float*)d_in[5];
  const float* hsr      = (const float*)d_in[6];
  const float* w1       = (const float*)d_in[7];
  const float* b1       = (const float*)d_in[8];
  const float* w2       = (const float*)d_in[9];
  const float* b2       = (const float*)d_in[10];
  const float* w3       = (const float*)d_in[11];
  const float* b3       = (const float*)d_in[12];
  const float* w4       = (const float*)d_in[13];
  const float* b4       = (const float*)d_in[14];

  float* out     = (float*)d_out;
  float* density = out + kT;
  float* states  = out + kT + kN;
  float* sumdens = (float*)d_ws;

  const int tiles = (kN + 15) / 16;                 // 1269 WMMA tiles
  mlp_density_kernel<<<tiles, 32, 0, stream>>>(alpha, beta, w1, b1, w2, b2,
                                               w3, b3, w4, b4, density);
  density_sum_kernel<<<1, 1024, 0, stream>>>(density, sumdens);
  scan_kernel<<<(kN + 255) / 256, 256, 0, stream>>>(x, alpha, beta, init_raw,
                                                    states);
  finalize_kernel<<<kT, 256, 0, stream>>>(x, density, states, sumdens,
                                          msr, mor, hsr, out);
}
